// SynNLI_Model_59785944760595
// MI455X (gfx1250) — matrix operand order repaired and checked by reading, not compiled, not run on hardware
//
#include <hip/hip_runtime.h>
#include <hip/hip_bf16.h>
#include <math.h>

// ============================================================================
// SynNLI model on MI455X (gfx1250, wave32, WMMA).
// All heavy GEMMs use v_wmma_f32_16x16x32_bf16 (fp32 accumulate), 64x64 block
// tiles, K-step 64, staged into LDS with global_load_async_to_lds_b128
// (ASYNCcnt) on the full-tile fast path.
// Irregular graph ops (segment max/sum, scatter) use fp32 global atomics.
// ============================================================================

#define DM    512      // model dim
#define NHEAD 8
#define DHEAD 64
#define LPMAX 384      // len_p in [128,385)
#define LHMAX 96       // len_h in [32,97)
#define LDA   72       // LDS A-tile row stride (elems): 144B, 16B-aligned
#define LDB   72       // LDS B-tile row stride

typedef __bf16 bf16;
typedef __attribute__((ext_vector_type(16))) __bf16 v16bf;
typedef __attribute__((ext_vector_type(8)))  float  v8f;

// ---------------------------------------------------------------- bf16 utils
__device__ __forceinline__ bf16 f2bf(float f) {
  union { float f; unsigned u; } v; v.f = f;
  unsigned r = v.u + 0x7FFFu + ((v.u >> 16) & 1u);   // round-to-nearest-even
  unsigned short h = (unsigned short)(r >> 16);
  bf16 out;
  __builtin_memcpy(&out, &h, 2);
  return out;
}
__device__ __forceinline__ float bf2f(bf16 b) {
  unsigned short h; __builtin_memcpy(&h, &b, 2);
  union { unsigned u; float f; } v; v.u = ((unsigned)h) << 16;
  return v.f;
}

__device__ __forceinline__ void atomicMaxFloat(float* addr, float v) {
  // monotone int trick: works with init = -1e30
  if (v >= 0.f) atomicMax((int*)addr, __float_as_int(v));
  else          atomicMin((unsigned int*)addr, __float_as_uint(v));
}

// ------------------------------------------------- CDNA5 async copy to LDS
// GLOBAL_LOAD_ASYNC_TO_LDS_B128 (GV mode): per-lane 16B DMA global -> LDS,
// tracked with ASYNCcnt (CDNA5 ISA). VDST = LDS byte address
// (low 32 bits of the flat shared pointer), VADDR = 64-bit global address.
__device__ __forceinline__ void async_b128(unsigned lds_off, unsigned long long ga) {
  asm volatile("global_load_async_to_lds_b128 %0, %1, off"
               :: "v"(lds_off), "v"(ga) : "memory");
}
__device__ __forceinline__ void async_wait() {
  asm volatile("s_wait_asynccnt 0" ::: "memory");
}

// ------------------------------------------------------------ fill / convert
__global__ void k_fill_f32(float* p, float v, long n) {
  long i = (long)blockIdx.x * blockDim.x + threadIdx.x;
  long s = (long)gridDim.x * blockDim.x;
  for (; i < n; i += s) p[i] = v;
}
__global__ void k_fill_bf16(bf16* p, float v, long n) {
  long i = (long)blockIdx.x * blockDim.x + threadIdx.x;
  long s = (long)gridDim.x * blockDim.x;
  bf16 bv = f2bf(v);
  for (; i < n; i += s) p[i] = bv;
}
__global__ void k_fill_i32(int* p, int v, long n) {
  long i = (long)blockIdx.x * blockDim.x + threadIdx.x;
  long s = (long)gridDim.x * blockDim.x;
  for (; i < n; i += s) p[i] = v;
}
__global__ void k_f32_to_bf16(const float* src, bf16* dst, long n) {
  long i = (long)blockIdx.x * blockDim.x + threadIdx.x;
  long s = (long)gridDim.x * blockDim.x;
  for (; i < n; i += s) dst[i] = f2bf(src[i]);
}

// ------------------------------------------------------------ embedding gather
__global__ void k_gather(const float* __restrict__ tab, const int* __restrict__ idx,
                         float* __restrict__ hf, bf16* __restrict__ hb, int n) {
  int node = blockIdx.x;
  if (node >= n) return;
  const float* src = tab + (size_t)idx[node] * DM;
  float* df = hf + (size_t)node * DM;
  bf16*  db = hb + (size_t)node * DM;
  for (int t = threadIdx.x; t < DM; t += blockDim.x) {
    float v = src[t]; df[t] = v; db[t] = f2bf(v);
  }
}

// ============================================================================
// WMMA GEMM:  C[b] = A[b] (MxK, bf16, row-major) @ B[b] (KxN, bf16, row-major)
// Block = 256 threads = 8 waves; block tile 64x64; wave tile 16x32; K-step 64
// (4 WMMAs per wave per staging round).
// CDNA5 fragment layouts per ISA 7.12.2:
//   A (16x32 bf16): lane = M (mod 16); elems 0..7 -> K = 8*(lane>=16)+e,
//                   elems 8..15 -> K = 16 + 8*(lane>=16) + (e-8)
//   B (32x16 bf16): lane = N (mod 16); elems follow the same K order as A
//   C/D (16x16 f32): vgpr r -> M = r + 8*(lane>=16); N = lane & 15
// ============================================================================
__device__ __forceinline__ v16bf frag_a(const bf16* As, int mbase, int lane, int kk) {
  v16bf a;
  const bf16* row = As + (mbase + (lane & 15)) * LDA + kk;
  int hi = (lane >> 4) * 8;
#pragma unroll
  for (int e = 0; e < 8; ++e) { a[e] = row[hi + e]; a[e + 8] = row[16 + hi + e]; }
  return a;
}
__device__ __forceinline__ v16bf frag_b(const bf16* Bs, int nbase, int lane, int kk) {
  v16bf b;
  const bf16* col = Bs + kk * LDB + nbase + (lane & 15);   // lane = n
  int hi = (lane >> 4) * 8;
#pragma unroll
  for (int e = 0; e < 8; ++e) { b[e] = col[(hi + e) * LDB]; b[e + 8] = col[(16 + hi + e) * LDB]; }
  return b;
}
__device__ __forceinline__ v16bf frag_bt(const bf16* Bt, int nbase, int lane, int kk) {
  v16bf b;                                             // tile stored [n][k]; lane = n
  const bf16* row = Bt + (nbase + (lane & 15)) * LDB + kk;
  int hi = (lane >> 4) * 8;
#pragma unroll
  for (int e = 0; e < 8; ++e) { b[e] = row[hi + e]; b[e + 8] = row[16 + hi + e]; }
  return b;
}

__global__ __launch_bounds__(256)
void k_gemm_bf16(const bf16* __restrict__ A, long sA,
                 const bf16* __restrict__ B, long sB,
                 const float* __restrict__ bias,
                 float* __restrict__ Cf, bf16* __restrict__ Cb, long sC,
                 int M, int N, int K, int relu) {
  __shared__ bf16 As[64 * LDA];
  __shared__ bf16 Bs[64 * LDB];
  int bz = blockIdx.z;
  const bf16* Ab = A + (size_t)bz * sA;
  const bf16* Bb = B + (size_t)bz * sB;
  int m0 = blockIdx.y * 64, n0 = blockIdx.x * 64;
  int tid = threadIdx.x, lane = tid & 31, wave = tid >> 5;
  int wm = (wave & 3) * 16, wn = (wave >> 2) * 32;
  v8f acc[2];
  acc[0] = (v8f){0.f, 0.f, 0.f, 0.f, 0.f, 0.f, 0.f, 0.f};
  acc[1] = (v8f){0.f, 0.f, 0.f, 0.f, 0.f, 0.f, 0.f, 0.f};

  int ar = tid >> 2, ac = (tid & 3) * 16;     // staging coords: 64 rows x 64 cols
  bool afull = (m0 + 64 <= M);                // block-uniform: async fast path
  bool bfull = ((N & 7) == 0) && (n0 + 64 <= N);
  unsigned lda_off = (unsigned)(size_t)(void*)&As[ar * LDA + ac];
  unsigned ldb_off = (unsigned)(size_t)(void*)&Bs[ar * LDB + ac];

  for (int k0 = 0; k0 < K; k0 += 64) {
    // ---- stage A tile (64 rows x 64 k) ----
    if (afull) {
      unsigned long long ga =
          (unsigned long long)(size_t)(Ab + (size_t)(m0 + ar) * K + k0 + ac);
      async_b128(lda_off, ga);
      async_b128(lda_off + 16, ga + 16);
    } else {
      uint4 v0 = {0u, 0u, 0u, 0u}, v1 = {0u, 0u, 0u, 0u};
      if (m0 + ar < M) {
        const bf16* src = Ab + (size_t)(m0 + ar) * K + k0 + ac;
        v0 = *(const uint4*)src;
        v1 = *(const uint4*)(src + 8);
      }
      *(uint4*)(&As[ar * LDA + ac]) = v0;
      *(uint4*)(&As[ar * LDA + ac + 8]) = v1;
    }
    // ---- stage B tile (64 k x 64 n) ----
    if (bfull) {
      unsigned long long gb =
          (unsigned long long)(size_t)(Bb + (size_t)(k0 + ar) * N + n0 + ac);
      async_b128(ldb_off, gb);
      async_b128(ldb_off + 16, gb + 16);
    } else {
#pragma unroll
      for (int j = 0; j < 16; ++j) {
        bf16 v = f2bf(0.f);
        if (n0 + ac + j < N) v = Bb[(size_t)(k0 + ar) * N + n0 + ac + j];
        Bs[ar * LDB + ac + j] = v;
      }
    }
    if (afull || bfull) async_wait();
    __syncthreads();
    if (k0 + 64 < K)  // global_prefetch_b8 for next K tile
      __builtin_prefetch(Ab + (size_t)(m0 + ar) * K + k0 + 64, 0, 1);
#pragma unroll
    for (int kk = 0; kk < 64; kk += 32) {
      v16bf fa  = frag_a(As, wm, lane, kk);
      v16bf fb0 = frag_b(Bs, wn, lane, kk);
      v16bf fb1 = frag_b(Bs, wn + 16, lane, kk);
      acc[0] = __builtin_amdgcn_wmma_f32_16x16x32_bf16(false, fa, false, fb0,
                                                       (short)0, acc[0], false, false);
      acc[1] = __builtin_amdgcn_wmma_f32_16x16x32_bf16(false, fa, false, fb1,
                                                       (short)0, acc[1], false, false);
    }
    __syncthreads();
  }

  int rbase = m0 + wm + ((lane >> 4) * 8);
#pragma unroll
  for (int g = 0; g < 2; ++g) {
    int col = n0 + wn + g * 16 + (lane & 15);
    if (col < N) {
      float bv = bias ? bias[col] : 0.f;
#pragma unroll
      for (int r = 0; r < 8; ++r) {
        int row = rbase + r;
        if (row < M) {
          float v = acc[g][r] + bv;
          if (relu && v < 0.f) v = 0.f;
          size_t off = (size_t)bz * sC + (size_t)row * N + col;
          if (Cf) Cf[off] = v;
          if (Cb) Cb[off] = f2bf(v);
        }
      }
    }
  }
}

// C[b] = A[b] (MxK) @ B[b] (NxK)^T  -> (MxN), used for E = Q K^T
__global__ __launch_bounds__(256)
void k_gemm_bt_bf16(const bf16* __restrict__ A, long sA,
                    const bf16* __restrict__ B, long sB,
                    float* __restrict__ Cf, long sC, int M, int N, int K) {
  __shared__ bf16 As[64 * LDA];
  __shared__ bf16 Bt[64 * LDB];   // [n][k]
  int bz = blockIdx.z;
  const bf16* Ab = A + (size_t)bz * sA;
  const bf16* Bb = B + (size_t)bz * sB;
  int m0 = blockIdx.y * 64, n0 = blockIdx.x * 64;
  int tid = threadIdx.x, lane = tid & 31, wave = tid >> 5;
  int wm = (wave & 3) * 16, wn = (wave >> 2) * 32;
  v8f acc[2];
  acc[0] = (v8f){0.f, 0.f, 0.f, 0.f, 0.f, 0.f, 0.f, 0.f};
  acc[1] = (v8f){0.f, 0.f, 0.f, 0.f, 0.f, 0.f, 0.f, 0.f};

  int ar = tid >> 2, ac = (tid & 3) * 16;
  bool afull = (m0 + 64 <= M);
  bool bfull = (n0 + 64 <= N);               // K always a multiple of 64 here
  unsigned lda_off = (unsigned)(size_t)(void*)&As[ar * LDA + ac];
  unsigned ldb_off = (unsigned)(size_t)(void*)&Bt[ar * LDB + ac];

  for (int k0 = 0; k0 < K; k0 += 64) {
    if (afull) {
      unsigned long long ga =
          (unsigned long long)(size_t)(Ab + (size_t)(m0 + ar) * K + k0 + ac);
      async_b128(lda_off, ga);
      async_b128(lda_off + 16, ga + 16);
    } else {
      uint4 v0 = {0u, 0u, 0u, 0u}, v1 = {0u, 0u, 0u, 0u};
      if (m0 + ar < M) {
        const bf16* src = Ab + (size_t)(m0 + ar) * K + k0 + ac;
        v0 = *(const uint4*)src;
        v1 = *(const uint4*)(src + 8);
      }
      *(uint4*)(&As[ar * LDA + ac]) = v0;
      *(uint4*)(&As[ar * LDA + ac + 8]) = v1;
    }
    if (bfull) {
      unsigned long long gb =
          (unsigned long long)(size_t)(Bb + (size_t)(n0 + ar) * K + k0 + ac);
      async_b128(ldb_off, gb);
      async_b128(ldb_off + 16, gb + 16);
    } else {
      uint4 v0 = {0u, 0u, 0u, 0u}, v1 = {0u, 0u, 0u, 0u};
      if (n0 + ar < N) {
        const bf16* src = Bb + (size_t)(n0 + ar) * K + k0 + ac;
        v0 = *(const uint4*)src;
        v1 = *(const uint4*)(src + 8);
      }
      *(uint4*)(&Bt[ar * LDB + ac]) = v0;
      *(uint4*)(&Bt[ar * LDB + ac + 8]) = v1;
    }
    if (afull || bfull) async_wait();
    __syncthreads();
#pragma unroll
    for (int kk = 0; kk < 64; kk += 32) {
      v16bf fa  = frag_a(As, wm, lane, kk);
      v16bf fb0 = frag_bt(Bt, wn, lane, kk);
      v16bf fb1 = frag_bt(Bt, wn + 16, lane, kk);
      acc[0] = __builtin_amdgcn_wmma_f32_16x16x32_bf16(false, fa, false, fb0,
                                                       (short)0, acc[0], false, false);
      acc[1] = __builtin_amdgcn_wmma_f32_16x16x32_bf16(false, fa, false, fb1,
                                                       (short)0, acc[1], false, false);
    }
    __syncthreads();
  }
  int rbase = m0 + wm + ((lane >> 4) * 8);
#pragma unroll
  for (int g = 0; g < 2; ++g) {
    int col = n0 + wn + g * 16 + (lane & 15);
    if (col < N) {
#pragma unroll
      for (int r = 0; r < 8; ++r) {
        int row = rbase + r;
        if (row < M) Cf[(size_t)bz * sC + (size_t)row * N + col] = acc[g][r];
      }
    }
  }
}

// ------------------------------------------------------------------ GAT parts
// per (node, head): a_s = <xw[n,h,:], att_src[h,:]>, a_d likewise. wave per pair.
__global__ void k_gat_att(const float* __restrict__ xw,
                          const float* __restrict__ att_src,
                          const float* __restrict__ att_dst,
                          float* __restrict__ as_, float* __restrict__ ad_, int N) {
  long idx = (long)blockIdx.x * 8 + (threadIdx.x >> 5);
  int lane = threadIdx.x & 31;
  if (idx >= (long)N * NHEAD) return;
  int n = (int)(idx >> 3), h = (int)(idx & 7);
  const float* xv = xw + (size_t)n * DM + h * DHEAD;
  float s1 = xv[lane] * att_src[h * DHEAD + lane] + xv[lane + 32] * att_src[h * DHEAD + lane + 32];
  float s2 = xv[lane] * att_dst[h * DHEAD + lane] + xv[lane + 32] * att_dst[h * DHEAD + lane + 32];
  for (int o = 16; o; o >>= 1) { s1 += __shfl_xor(s1, o, 32); s2 += __shfl_xor(s2, o, 32); }
  if (lane == 0) { as_[idx] = s1; ad_[idx] = s2; }
}

__global__ void k_edge_logit(const int* __restrict__ ei, int E0, long Etot,
                             const float* __restrict__ as_, const float* __restrict__ ad_,
                             float* __restrict__ ev, float* __restrict__ mseg) {
  long t = (long)blockIdx.x * 256 + threadIdx.x;
  if (t >= Etot * NHEAD) return;
  long e = t >> 3; int h = (int)(t & 7);
  int s, d;
  if (e < E0) { s = ei[e]; d = ei[E0 + e]; } else { s = d = (int)(e - E0); }
  float v = as_[(size_t)s * NHEAD + h] + ad_[(size_t)d * NHEAD + h];
  if (v < 0.f) v *= 0.2f;                       // leaky_relu, slope 0.2
  ev[t] = v;
  atomicMaxFloat(&mseg[(size_t)d * NHEAD + h], v);
}

__global__ void k_edge_exp(const int* __restrict__ ei, int E0, long Etot,
                           float* __restrict__ ev, const float* __restrict__ mseg,
                           float* __restrict__ den) {
  long t = (long)blockIdx.x * 256 + threadIdx.x;
  if (t >= Etot * NHEAD) return;
  long e = t >> 3; int h = (int)(t & 7);
  int d;
  if (e < E0) d = ei[E0 + e]; else d = (int)(e - E0);
  float ex = __expf(ev[t] - mseg[(size_t)d * NHEAD + h]);
  ev[t] = ex;
  atomicAdd(&den[(size_t)d * NHEAD + h], ex);
}

// wave per (edge, head): acc[dst,h,:] += alpha * xw[src,h,:]
__global__ void k_edge_scatter(const int* __restrict__ ei, int E0, long Etot,
                               const float* __restrict__ ev, const float* __restrict__ den,
                               const float* __restrict__ xw, float* __restrict__ acc) {
  long idx = (long)blockIdx.x * 8 + (threadIdx.x >> 5);
  int lane = threadIdx.x & 31;
  if (idx >= Etot * NHEAD) return;
  long e = idx >> 3; int h = (int)(idx & 7);
  int s, d;
  if (e < E0) { s = ei[e]; d = ei[E0 + e]; } else { s = d = (int)(e - E0); }
  float alpha = ev[idx] / (den[(size_t)d * NHEAD + h] + 1e-16f);
  const float* xv = xw + (size_t)s * DM + h * DHEAD;
  float* av = acc + (size_t)d * DM + h * DHEAD;
  atomicAdd(&av[lane],      alpha * xv[lane]);
  atomicAdd(&av[lane + 32], alpha * xv[lane + 32]);
}

__global__ void k_bias_out(const float* __restrict__ acc, const float* __restrict__ bias,
                           float* __restrict__ hf, bf16* __restrict__ hb, long n) {
  long i = (long)blockIdx.x * 256 + threadIdx.x;
  if (i >= n) return;
  float v = acc[i] + bias[i & (DM - 1)];
  hf[i] = v; hb[i] = f2bf(v);
}

// ------------------------------------------------------------------- padding
__global__ void k_count(const int* __restrict__ batch, int* __restrict__ counts, int n) {
  int i = blockIdx.x * 256 + threadIdx.x;
  if (i < n) atomicAdd(&counts[batch[i]], 1);
}
__global__ void k_scan(const int* __restrict__ counts, int* __restrict__ starts, int B) {
  __shared__ int s[256];
  int t = threadIdx.x;
  s[t] = (t < B) ? counts[t] : 0;
  __syncthreads();
  for (int off = 1; off < 256; off <<= 1) {
    int v = (t >= off) ? s[t - off] : 0;
    __syncthreads();
    s[t] += v;
    __syncthreads();
  }
  if (t < B) starts[t] = s[t] - counts[t];
}
__global__ void k_pad(const float* __restrict__ h, const int* __restrict__ batch,
                      const int* __restrict__ starts, float* __restrict__ bpf,
                      bf16* __restrict__ bpb, float* __restrict__ mask,
                      int Lmax, int Nnodes) {
  int i = blockIdx.x;
  if (i >= Nnodes) return;
  int b = batch[i], pos = i - starts[b];
  if (pos >= Lmax) return;
  size_t off = ((size_t)b * Lmax + pos) * DM;
  for (int t = threadIdx.x; t < DM; t += blockDim.x) {
    float v = h[(size_t)i * DM + t];
    bpf[off + t] = v; bpb[off + t] = f2bf(v);
  }
  if (threadIdx.x == 0) mask[b * Lmax + pos] = 1.f;
}

// ------------------------------------------------------------ masked softmax
__global__ __launch_bounds__(128)
void k_softmax(const float* __restrict__ E, const float* __restrict__ maskh,
               const float* __restrict__ maskp, bf16* __restrict__ alpha,
               const int* __restrict__ pLp, const int* __restrict__ pLh,
               float inv_sqrt_d) {
  int n = blockIdx.x, b = blockIdx.y;
  int Lp = *pLp; if (Lp > LPMAX) Lp = LPMAX;
  int Lh = *pLh; if (Lh > LHMAX) Lh = LHMAX;
  const float* e = E + ((size_t)b * LHMAX + n) * LPMAX;
  bf16* al = alpha + ((size_t)b * LHMAX + n) * LPMAX;
  int t = threadIdx.x;
  if (n >= Lh) {           // rows outside the actual padded tensor: zero probs
    for (int m = t; m < LPMAX; m += 128) al[m] = f2bf(0.f);
    return;
  }
  float mh = maskh[b * LHMAX + n];
  float z[3], mx = -3.0e38f;
#pragma unroll
  for (int j = 0; j < 3; ++j) {
    int m = t + j * 128;
    float v = -3.0e38f;
    if (m < Lp) {
      float msk = mh * maskp[b * LPMAX + m];
      float ev = (msk == 0.f) ? -1e10f : e[m];
      v = ev * inv_sqrt_d;
    }
    z[j] = v;
    mx = fmaxf(mx, v);
  }
  __shared__ float red[4];
  for (int o = 16; o; o >>= 1) mx = fmaxf(mx, __shfl_xor(mx, o, 32));
  if ((t & 31) == 0) red[t >> 5] = mx;
  __syncthreads();
  mx = fmaxf(fmaxf(red[0], red[1]), fmaxf(red[2], red[3]));
  __syncthreads();
  float ex[3], sm = 0.f;
#pragma unroll
  for (int j = 0; j < 3; ++j) {
    int m = t + j * 128;
    float v = (m < Lp) ? __expf(z[j] - mx) : 0.f;
    ex[j] = v; sm += v;
  }
  for (int o = 16; o; o >>= 1) sm += __shfl_xor(sm, o, 32);
  if ((t & 31) == 0) red[t >> 5] = sm;
  __syncthreads();
  sm = red[0] + red[1] + red[2] + red[3];
  float inv = 1.f / sm;
#pragma unroll
  for (int j = 0; j < 3; ++j) {
    int m = t + j * 128;
    if (m < LPMAX) al[m] = f2bf((m < Lp) ? ex[j] * inv : 0.f);
  }
}

// ----------------------------------------------------- compare-feature build
__global__ void k_feat(const float* __restrict__ phat, const float* __restrict__ bh,
                       bf16* __restrict__ feat, long rows) {
  long r = blockIdx.x;
  if (r >= rows) return;
  const float* p = phat + r * DM;
  const float* h = bh + r * DM;
  bf16* f = feat + r * (4 * DM);
  for (int d = threadIdx.x; d < DM; d += blockDim.x) {
    float a = p[d], c = h[d];
    f[d] = f2bf(a);
    f[DM + d] = f2bf(c);
    f[2 * DM + d] = f2bf(a - c);
    f[3 * DM + d] = f2bf(a * c);
  }
}

// --------------------------------------------------------------- max/mean pool
__global__ void k_pool(const float* __restrict__ cmp, const int* __restrict__ pLh,
                       float* __restrict__ pooled, bf16* __restrict__ pooledb) {
  int b = blockIdx.x;
  int Lh = *pLh; if (Lh > LHMAX) Lh = LHMAX; if (Lh < 1) Lh = 1;
  for (int d = threadIdx.x; d < DM; d += blockDim.x) {
    float mx = -3.0e38f, sm = 0.f;
    for (int n = 0; n < Lh; ++n) {
      float v = cmp[((size_t)b * LHMAX + n) * DM + d];
      mx = v > mx ? v : mx;
      sm += v;
    }
    float mean = sm / (float)Lh;
    pooled[(size_t)b * 1024 + d] = mx;
    pooled[(size_t)b * 1024 + DM + d] = mean;
    pooledb[(size_t)b * 1024 + d] = f2bf(mx);
    pooledb[(size_t)b * 1024 + DM + d] = f2bf(mean);
  }
}

// ------------------------------------------------------------------ BCE loss
__global__ void k_loss(const float* __restrict__ logits, const float* __restrict__ label,
                       float* __restrict__ out, int n) {
  int i = blockIdx.x * 256 + threadIdx.x;
  if (i >= n) return;
  float x = logits[i], y = label[i];
  float l = fmaxf(x, 0.f) - x * y + log1pf(__expf(-fabsf(x)));
  atomicAdd(out, l / (float)n);
}

// ============================================================================
// Host orchestration
// ============================================================================
extern "C" void kernel_launch(void* const* d_in, const int* in_sizes, int n_in,
                              void* d_out, int out_size, void* d_ws, size_t ws_size,
                              hipStream_t stream) {
  const float* emb     = (const float*)d_in[0];
  const float* Wgat    = (const float*)d_in[1];
  const float* att_src = (const float*)d_in[2];
  const float* att_dst = (const float*)d_in[3];
  const float* gbias   = (const float*)d_in[4];
  const float* Wq      = (const float*)d_in[5];
  const float* Wk      = (const float*)d_in[6];
  const float* Wv      = (const float*)d_in[7];
  const float* w1      = (const float*)d_in[8];
  const float* b1      = (const float*)d_in[9];
  const float* w2      = (const float*)d_in[10];
  const float* b2      = (const float*)d_in[11];
  const float* c1      = (const float*)d_in[12];
  const float* cb1     = (const float*)d_in[13];
  const float* c2      = (const float*)d_in[14];
  const float* cb2     = (const float*)d_in[15];
  const float* label   = (const float*)d_in[16];
  const int* x_p       = (const int*)d_in[17];
  const int* x_h       = (const int*)d_in[18];
  const int* ei_p      = (const int*)d_in[19];
  const int* ei_h      = (const int*)d_in[20];
  const int* batch_p   = (const int*)d_in[21];
  const int* batch_h   = (const int*)d_in[22];
  const int* pLp       = (const int*)d_in[24];   // device scalar max_len_p
  const int* pLh       = (const int*)d_in[25];   // device scalar max_len_h

  const int NP  = in_sizes[17];
  const int NH  = in_sizes[18];
  const int E0P = in_sizes[19] / 2;              // 2*Np real edges
  const int E0H = in_sizes[20] / 2;
  const long ETP = (long)E0P + NP;               // + self loops
  const long ETH = (long)E0H + NH;
  const int Bsz = in_sizes[16] / 3;              // label is [B,3]

  // ---- workspace carve ----
  char* wptr = (char*)d_ws;
  auto carve = [&](size_t bytes) -> void* {
    void* p = (void*)wptr;
    wptr += (bytes + 255) & ~(size_t)255;
    return p;
  };
  bf16* wgat_b = (bf16*)carve((size_t)DM * DM * 2);
  bf16* wq_b   = (bf16*)carve((size_t)DM * DM * 2);
  bf16* wk_b   = (bf16*)carve((size_t)DM * DM * 2);
  bf16* wv_b   = (bf16*)carve((size_t)DM * DM * 2);
  bf16* w1_b   = (bf16*)carve((size_t)4 * DM * DM * 2);
  bf16* w2_b   = (bf16*)carve((size_t)DM * DM * 2);
  bf16* c1_b   = (bf16*)carve((size_t)2 * DM * DM * 2);
  bf16* c2_b   = (bf16*)carve((size_t)DM * 3 * 2);

  float* hp_f  = (float*)carve((size_t)NP * DM * 4);
  bf16*  hp_b  = (bf16*) carve((size_t)NP * DM * 2);
  float* hh_f  = (float*)carve((size_t)NH * DM * 4);
  bf16*  hh_b  = (bf16*) carve((size_t)NH * DM * 2);
  float* xw    = (float*)carve((size_t)NP * DM * 4);   // shared scratch (NP >= NH)
  float* gacc  = (float*)carve((size_t)NP * DM * 4);
  float* as_   = (float*)carve((size_t)NP * NHEAD * 4);
  float* ad_   = (float*)carve((size_t)NP * NHEAD * 4);
  float* mseg  = (float*)carve((size_t)NP * NHEAD * 4);
  float* den   = (float*)carve((size_t)NP * NHEAD * 4);
  float* ev    = (float*)carve((size_t)ETP * NHEAD * 4);
  int*   counts = (int*)carve(256 * 4);
  int*   starts = (int*)carve(256 * 4);

  float* bpf   = (float*)carve((size_t)Bsz * LPMAX * DM * 4);
  bf16*  bpb   = (bf16*) carve((size_t)Bsz * LPMAX * DM * 2);
  float* maskp = (float*)carve((size_t)Bsz * LPMAX * 4);
  float* bhf   = (float*)carve((size_t)Bsz * LHMAX * DM * 4);
  bf16*  bhb   = (bf16*) carve((size_t)Bsz * LHMAX * DM * 2);
  float* maskh = (float*)carve((size_t)Bsz * LHMAX * 4);

  bf16*  Qb    = (bf16*) carve((size_t)Bsz * LHMAX * DM * 2);
  bf16*  Kb    = (bf16*) carve((size_t)Bsz * LPMAX * DM * 2);
  bf16*  Vb    = (bf16*) carve((size_t)Bsz * LPMAX * DM * 2);
  float* Ef    = (float*)carve((size_t)Bsz * LHMAX * LPMAX * 4);
  bf16*  alb   = (bf16*) carve((size_t)Bsz * LHMAX * LPMAX * 2);
  float* phat  = (float*)carve((size_t)Bsz * LHMAX * DM * 4);
  bf16*  featb = (bf16*) carve((size_t)Bsz * LHMAX * 4 * DM * 2);
  bf16*  cmp1b = (bf16*) carve((size_t)Bsz * LHMAX * DM * 2);
  float* cmpf  = (float*)carve((size_t)Bsz * LHMAX * DM * 4);
  float* pooled  = (float*)carve((size_t)Bsz * 1024 * 4);
  bf16*  pooledb = (bf16*) carve((size_t)Bsz * 1024 * 2);
  bf16*  g1b     = (bf16*) carve((size_t)Bsz * DM * 2);
  (void)ws_size; (void)n_in; (void)out_size;

  float* loss_ptr   = (float*)d_out;
  float* logits_ptr = (float*)d_out + 1;

  // ---- small launch helpers ----
  auto fgrid = [](long n) -> unsigned {
    long g = (n + 255) / 256;
    return (unsigned)(g > 8192 ? 8192 : g);
  };
  auto fill_f = [&](float* p, float v, long n) {
    k_fill_f32<<<fgrid(n), 256, 0, stream>>>(p, v, n);
  };
  auto fill_b = [&](bf16* p, float v, long n) {
    k_fill_bf16<<<fgrid(n), 256, 0, stream>>>(p, v, n);
  };
  auto cvt = [&](const float* s, bf16* d, long n) {
    k_f32_to_bf16<<<fgrid(n), 256, 0, stream>>>(s, d, n);
  };
  auto gemm = [&](const bf16* A, long sA, const bf16* Bw, long sB, const float* bias,
                  float* Cf, bf16* Cb, long sC, int M, int N, int K, int relu, int batch) {
    dim3 g((unsigned)((N + 63) / 64), (unsigned)((M + 63) / 64), (unsigned)batch);
    k_gemm_bf16<<<g, 256, 0, stream>>>(A, sA, Bw, sB, bias, Cf, Cb, sC, M, N, K, relu);
  };

  // ---- weight conversions (deterministic, re-done every call) ----
  cvt(Wgat, wgat_b, (long)DM * DM);
  cvt(Wq, wq_b, (long)DM * DM);
  cvt(Wk, wk_b, (long)DM * DM);
  cvt(Wv, wv_b, (long)DM * DM);
  cvt(w1, w1_b, (long)4 * DM * DM);
  cvt(w2, w2_b, (long)DM * DM);
  cvt(c1, c1_b, (long)2 * DM * DM);
  cvt(c2, c2_b, (long)DM * 3);

  // ---- embedding gather ----
  k_gather<<<NP, 128, 0, stream>>>(emb, x_p, hp_f, hp_b, NP);
  k_gather<<<NH, 128, 0, stream>>>(emb, x_h, hh_f, hh_b, NH);

  // ---- 3-layer GAT (shared weights) ----
  auto run_gat = [&](float* hf, bf16* hb, int Nn, const int* ei, int E0, long Etot) {
    for (int layer = 0; layer < 3; ++layer) {
      gemm(hb, 0, wgat_b, 0, nullptr, xw, nullptr, 0, Nn, DM, DM, 0, 1);   // xw = h @ W
      k_gat_att<<<(unsigned)Nn, 256, 0, stream>>>(xw, att_src, att_dst, as_, ad_, Nn);
      fill_f(mseg, -1e30f, (long)Nn * NHEAD);
      fill_f(den, 0.f, (long)Nn * NHEAD);
      fill_f(gacc, 0.f, (long)Nn * DM);
      long tE = Etot * NHEAD;
      k_edge_logit<<<(unsigned)((tE + 255) / 256), 256, 0, stream>>>(ei, E0, Etot, as_, ad_, ev, mseg);
      k_edge_exp<<<(unsigned)((tE + 255) / 256), 256, 0, stream>>>(ei, E0, Etot, ev, mseg, den);
      k_edge_scatter<<<(unsigned)Etot, 256, 0, stream>>>(ei, E0, Etot, ev, den, xw, gacc);
      k_bias_out<<<(unsigned)(((long)Nn * DM + 255) / 256), 256, 0, stream>>>(gacc, gbias, hf, hb, (long)Nn * DM);
    }
  };
  run_gat(hp_f, hp_b, NP, ei_p, E0P, ETP);
  run_gat(hh_f, hh_b, NH, ei_h, E0H, ETH);

  // ---- pad to [B, L, D] ----
  auto pad = [&](const float* hf, const int* batch, int Nn, int Lmax,
                 float* bf_, bf16* bb_, float* msk) {
    fill_f(bf_, 0.f, (long)Bsz * Lmax * DM);
    fill_b(bb_, 0.f, (long)Bsz * Lmax * DM);
    fill_f(msk, 0.f, (long)Bsz * Lmax);
    k_fill_i32<<<1, 256, 0, stream>>>(counts, 0, 256);
    k_count<<<(unsigned)((Nn + 255) / 256), 256, 0, stream>>>(batch, counts, Nn);
    k_scan<<<1, 256, 0, stream>>>(counts, starts, Bsz);
    k_pad<<<(unsigned)Nn, 128, 0, stream>>>(hf, batch, starts, bf_, bb_, msk, Lmax, Nn);
  };
  pad(hp_f, batch_p, NP, LPMAX, bpf, bpb, maskp);
  pad(hh_f, batch_h, NH, LHMAX, bhf, bhb, maskh);

  // ---- cross attention ----
  gemm(bhb, 0, wq_b, 0, nullptr, nullptr, Qb, 0, Bsz * LHMAX, DM, DM, 0, 1);  // Q
  gemm(bpb, 0, wk_b, 0, nullptr, nullptr, Kb, 0, Bsz * LPMAX, DM, DM, 0, 1);  // K
  gemm(bpb, 0, wv_b, 0, nullptr, nullptr, Vb, 0, Bsz * LPMAX, DM, DM, 0, 1);  // V

  {   // E[b] = Q[b] @ K[b]^T
    dim3 g((unsigned)((LPMAX + 63) / 64), (unsigned)((LHMAX + 63) / 64), (unsigned)Bsz);
    k_gemm_bt_bf16<<<g, 256, 0, stream>>>(Qb, (long)LHMAX * DM, Kb, (long)LPMAX * DM,
                                          Ef, (long)LHMAX * LPMAX, LHMAX, LPMAX, DM);
  }
  {   // masked softmax over premise axis (scale 1/sqrt(D))
    dim3 g((unsigned)LHMAX, (unsigned)Bsz);
    k_softmax<<<g, 128, 0, stream>>>(Ef, maskh, maskp, alb, pLp, pLh,
                                     1.0f / sqrtf((float)DM));
  }
  // p_hat[b] = A[b] @ V[b]
  gemm(alb, (long)LHMAX * LPMAX, Vb, (long)LPMAX * DM, nullptr,
       phat, nullptr, (long)LHMAX * DM, LHMAX, DM, LPMAX, 0, Bsz);

  // ---- comparison MLP ----
  k_feat<<<(unsigned)(Bsz * LHMAX), 128, 0, stream>>>(phat, bhf, featb, (long)Bsz * LHMAX);
  gemm(featb, 0, w1_b, 0, b1, nullptr, cmp1b, 0, Bsz * LHMAX, DM, 4 * DM, 1, 1); // relu
  gemm(cmp1b, 0, w2_b, 0, b2, cmpf, nullptr, 0, Bsz * LHMAX, DM, DM, 0, 1);

  // ---- pooling + classifier ----
  k_pool<<<(unsigned)Bsz, 256, 0, stream>>>(cmpf, pLh, pooled, pooledb);
  gemm(pooledb, 0, c1_b, 0, cb1, nullptr, g1b, 0, Bsz, DM, 2 * DM, 1, 1);        // relu
  gemm(g1b, 0, c2_b, 0, cb2, logits_ptr, nullptr, 0, Bsz, 3, DM, 0, 1);

  // ---- BCE-with-logits loss (mean) ----
  fill_f(loss_ptr, 0.f, 1);
  k_loss<<<(unsigned)((Bsz * 3 + 255) / 256), 256, 0, stream>>>(logits_ptr, label,
                                                                loss_ptr, Bsz * 3);
}
